// MoELayer_27908697489570
// MI455X (gfx1250) — compile-verified, matrix operand order checked by reading, not run on hardware
//
#include <hip/hip_runtime.h>
#include <math.h>

// ---------------------------------------------------------------------------
// MoE layer for MI455X (gfx1250), wave32 + WMMA bf16 path.
// ---------------------------------------------------------------------------

typedef __bf16 bf16;
typedef __attribute__((ext_vector_type(16))) __bf16        v16bf;
typedef __attribute__((ext_vector_type(8)))  float         v8f;
typedef __attribute__((ext_vector_type(4)))  unsigned int  u32x4;
typedef __attribute__((ext_vector_type(4)))  float         f32x4;

#define T_TOK 8192
#define HDIM  1024
#define FDIM  4096
#define NEXP  8
#define CAP   1280   // int(1.25 * 8192/8)

#define MT 128       // M tile per workgroup
#define NT 128       // N tile per workgroup
#define KT 32        // K depth per WMMA / LDS stage
#define LDS_STRIDE 40  // bf16 elems per LDS row (80B, 16B-aligned, bank-spread)

// Async global->LDS (CDNA5): use if the toolchain exposes the builtins.
#if defined(__has_builtin)
#if __has_builtin(__builtin_amdgcn_global_load_async_to_lds_b128) && \
    __has_builtin(__builtin_amdgcn_s_wait_asynccnt)
#define USE_ASYNC_LDS 1
#endif
#endif
#ifndef USE_ASYNC_LDS
#define USE_ASYNC_LDS 0
#endif

#if USE_ASYNC_LDS
// Builtin signature (from hipcc diagnostic): (v4i addrspace(1)*, v4i addrspace(3)*, imm, imm)
typedef __attribute__((__vector_size__(4 * sizeof(int)))) int v4i_t;
typedef __attribute__((address_space(1))) v4i_t* gvec_p;
typedef __attribute__((address_space(3))) v4i_t* lvec_p;
#endif

// Assemble a v16bf WMMA fragment from two contiguous 16B chunks in LDS.
// ISA 16-bit A/B layout: lane L -> (row = L&15), elements 0..7 cover
// k = (L>>4)*8 + 0..7, elements 8..15 cover k = 16 + (L>>4)*8 + 0..7.
union FragU { u32x4 q[2]; v16bf f; };
__device__ __forceinline__ v16bf load_frag(const bf16* p) {
  FragU u;
  u.q[0] = *(const u32x4*)(p);
  u.q[1] = *(const u32x4*)(p + 16);
  return u.f;
}

__device__ __forceinline__ unsigned pack_bf16(float a, float b) {
  union { bf16 h[2]; unsigned u; } cv;
  cv.h[0] = (bf16)a;
  cv.h[1] = (bf16)b;
  return cv.u;
}

__device__ __forceinline__ float gelu_exact(float v) {
  return 0.5f * v * (1.0f + erff(v * 0.70710678118654752f));
}

// ---------------------------------------------------------------------------
// 0) zero the scatter-add output
// ---------------------------------------------------------------------------
__global__ void moe_zero_kernel(f32x4* __restrict__ p, int n4) {
  int i = blockIdx.x * blockDim.x + threadIdx.x;
  if (i < n4) { f32x4 z = {0.f, 0.f, 0.f, 0.f}; p[i] = z; }
}

// ---------------------------------------------------------------------------
// 1) gating: logits = x @ gate_w, softmax -> probs (T x E). One wave / token.
// ---------------------------------------------------------------------------
__global__ __launch_bounds__(256)
void moe_gate_kernel(const float* __restrict__ x, const float* __restrict__ gw,
                     float* __restrict__ probs) {
  __shared__ float sgw[HDIM * NEXP];   // 32 KB
  for (int i = threadIdx.x; i < HDIM * NEXP; i += 256) sgw[i] = gw[i];
  __syncthreads();

  const int lane = threadIdx.x & 31, wid = threadIdx.x >> 5;
  const int t = blockIdx.x * 8 + wid;
  const float* xr = x + (size_t)t * HDIM;

  float acc[NEXP] = {0.f, 0.f, 0.f, 0.f, 0.f, 0.f, 0.f, 0.f};
  for (int k = lane; k < HDIM; k += 32) {
    float xv = xr[k];
#pragma unroll
    for (int e = 0; e < NEXP; ++e) acc[e] = fmaf(xv, sgw[k * NEXP + e], acc[e]);
  }
#pragma unroll
  for (int e = 0; e < NEXP; ++e) {
#pragma unroll
    for (int off = 16; off > 0; off >>= 1) acc[e] += __shfl_xor(acc[e], off, 32);
  }
  if (lane == 0) {
    float mx = acc[0];
#pragma unroll
    for (int e = 1; e < NEXP; ++e) mx = fmaxf(mx, acc[e]);
    float ex[NEXP], s = 0.f;
#pragma unroll
    for (int e = 0; e < NEXP; ++e) { ex[e] = expf(acc[e] - mx); s += ex[e]; }
    float inv = 1.f / s;
#pragma unroll
    for (int e = 0; e < NEXP; ++e) probs[(size_t)t * NEXP + e] = ex[e] * inv;
  }
}

// ---------------------------------------------------------------------------
// 2) per-expert top-CAP via full bitonic sort of 8192 packed keys in LDS.
//    key = orderable(score)<<32 | (0xFFFFFFFF - token)  (descending sort ->
//    higher score first, lower token index first on ties == lax.top_k).
// ---------------------------------------------------------------------------
__global__ __launch_bounds__(1024)
void moe_topk_kernel(const float* __restrict__ probs, int* __restrict__ idx,
                     float* __restrict__ scores, float* __restrict__ eload) {
  __shared__ unsigned long long keys[T_TOK];   // 64 KB
  const int e = blockIdx.x, tid = threadIdx.x;
  const int lane = tid & 31, wid = tid >> 5;

  for (int i = tid; i < T_TOK; i += 1024) {
    unsigned u = __float_as_uint(probs[(size_t)i * NEXP + e]);
    unsigned kb = (u & 0x80000000u) ? ~u : (u | 0x80000000u);
    keys[i] = ((unsigned long long)kb << 32) | (unsigned)(0xFFFFFFFFu - (unsigned)i);
  }
  __syncthreads();

  for (unsigned k = 2; k <= T_TOK; k <<= 1) {
    for (unsigned j = k >> 1; j > 0; j >>= 1) {
      for (int i = tid; i < T_TOK; i += 1024) {
        int l = i ^ (int)j;
        if (l > i) {
          unsigned long long a = keys[i], b = keys[l];
          bool up = ((i & k) == 0);
          bool sw = up ? (a < b) : (a > b);   // descending overall
          if (sw) { keys[i] = b; keys[l] = a; }
        }
      }
      __syncthreads();
    }
  }

  float lsum = 0.f;
  for (int c = tid; c < CAP; c += 1024) {
    unsigned long long kk = keys[c];
    unsigned tok = 0xFFFFFFFFu - (unsigned)(kk & 0xFFFFFFFFu);
    unsigned kb = (unsigned)(kk >> 32);
    unsigned u = (kb & 0x80000000u) ? (kb & 0x7FFFFFFFu) : ~kb;
    float sc = __uint_as_float(u);
    idx[e * CAP + c] = (int)tok;
    scores[e * CAP + c] = sc;
    lsum += sc;
  }
  __syncthreads();                    // done reading keys; reuse as scratch
#pragma unroll
  for (int off = 16; off > 0; off >>= 1) lsum += __shfl_xor(lsum, off, 32);
  if (lane == 0) ((float*)keys)[wid] = lsum;
  __syncthreads();
  if (tid < 32) {
    float v = ((float*)keys)[tid];
#pragma unroll
    for (int off = 16; off > 0; off >>= 1) v += __shfl_xor(v, off, 32);
    if (tid == 0) eload[e] = v / (float)T_TOK;
  }
}

// ---------------------------------------------------------------------------
// 3) lb_loss = var(expert_load, ddof=1) * E  -> out[T*H]
// ---------------------------------------------------------------------------
__global__ void moe_lbloss_kernel(const float* __restrict__ eload,
                                  float* __restrict__ out) {
  if (blockIdx.x == 0 && threadIdx.x == 0) {
    float m = 0.f;
#pragma unroll
    for (int e = 0; e < NEXP; ++e) m += eload[e];
    m *= (1.0f / NEXP);
    float v = 0.f;
#pragma unroll
    for (int e = 0; e < NEXP; ++e) { float d = eload[e] - m; v += d * d; }
    out[0] = (v / (float)(NEXP - 1)) * (float)NEXP;
  }
}

// ---------------------------------------------------------------------------
// 4) FFN1: h[e] = gelu(x[idx[e]] @ W1[e] + b1[e]), bf16 out.
//    M=CAP, N=FDIM, K=HDIM. 128x128 tile, 8 waves (4Mx2N), wave = 2x4 WMMA.
// ---------------------------------------------------------------------------
__global__ __launch_bounds__(256)
void moe_ffn1_kernel(const float* __restrict__ x, const float* __restrict__ W1,
                     const float* __restrict__ b1, const int* __restrict__ idx,
                     bf16* __restrict__ h) {
  __shared__ bf16 sA[2][MT][LDS_STRIDE];   // 20 KB
  __shared__ bf16 sB[2][NT][LDS_STRIDE];   // 20 KB  (B stored transposed [n][k])

  const int e  = blockIdx.z;
  const int m0 = blockIdx.y * MT;
  const int n0 = blockIdx.x * NT;
  const int tid = threadIdx.x;
  const int lane = tid & 31, wid = tid >> 5;
  const int wm = wid >> 1, wn = wid & 1;
  const int lmod = lane & 15, lhalf = lane >> 4;

  const int ar  = tid >> 1;               // staging row 0..127
  const int akq = (tid & 1) << 4;         // staging k offset 0/16
  const int tok = idx[e * CAP + m0 + ar];
  const float* aRow = x + (size_t)tok * HDIM;
  const float* w1e  = W1 + (size_t)e * HDIM * FDIM;

  v8f acc[2][4];
  v8f zf = {0.f, 0.f, 0.f, 0.f, 0.f, 0.f, 0.f, 0.f};
#pragma unroll
  for (int ms = 0; ms < 2; ++ms)
#pragma unroll
    for (int ns = 0; ns < 4; ++ns) acc[ms][ns] = zf;

  auto stage = [&](int k0, int buf) {
    // A: fp32 gather row -> v_cvt_pk bf16 pairs -> two b128 LDS stores
    f32x4 a4[4];
#pragma unroll
    for (int c = 0; c < 4; ++c) a4[c] = *(const f32x4*)(aRow + k0 + akq + 4 * c);
    unsigned pk[8];
#pragma unroll
    for (int c = 0; c < 4; ++c) {
      pk[c * 2 + 0] = pack_bf16(a4[c][0], a4[c][1]);
      pk[c * 2 + 1] = pack_bf16(a4[c][2], a4[c][3]);
    }
    u32x4 q0 = {pk[0], pk[1], pk[2], pk[3]};
    u32x4 q1 = {pk[4], pk[5], pk[6], pk[7]};
    *(u32x4*)&sA[buf][ar][akq]     = q0;
    *(u32x4*)&sA[buf][ar][akq + 8] = q1;
    // B: fp32 W1 [k][n] -> cvt bf16 -> LDS transposed [n][k]
#pragma unroll
    for (int it = 0; it < 4; ++it) {
      int v = tid + it * 256;
      int kk = v >> 5, nv = v & 31;
      f32x4 w = *(const f32x4*)(w1e + (size_t)(k0 + kk) * FDIM + n0 + nv * 4);
#pragma unroll
      for (int j = 0; j < 4; ++j) sB[buf][nv * 4 + j][kk] = (bf16)w[j];
    }
  };

  stage(0, 0);
  __syncthreads();
  const int KSTEPS = HDIM / KT;
  for (int ks = 0; ks < KSTEPS; ++ks) {
    const int buf = ks & 1;
    if (ks + 1 < KSTEPS) stage((ks + 1) * KT, buf ^ 1);
    v16bf af[2], bfr[4];
#pragma unroll
    for (int ms = 0; ms < 2; ++ms)
      af[ms] = load_frag(&sA[buf][wm * 32 + ms * 16 + lmod][lhalf * 8]);
#pragma unroll
    for (int ns = 0; ns < 4; ++ns)
      bfr[ns] = load_frag(&sB[buf][wn * 64 + ns * 16 + lmod][lhalf * 8]);
#pragma unroll
    for (int ms = 0; ms < 2; ++ms)
#pragma unroll
      for (int ns = 0; ns < 4; ++ns)
        acc[ms][ns] = __builtin_amdgcn_wmma_f32_16x16x32_bf16(
            false, af[ms], false, bfr[ns], (short)0, acc[ms][ns], false, false);
    __syncthreads();
  }

  // epilogue: + b1, exact GELU, bf16 store
  bf16* he = h + (size_t)e * CAP * FDIM;
  float bv[4];
#pragma unroll
  for (int ns = 0; ns < 4; ++ns)
    bv[ns] = b1[e * FDIM + n0 + wn * 64 + ns * 16 + lmod];
#pragma unroll
  for (int ms = 0; ms < 2; ++ms) {
#pragma unroll
    for (int r = 0; r < 8; ++r) {
      int m = m0 + wm * 32 + ms * 16 + lhalf * 8 + r;
      bf16* hrow = he + (size_t)m * FDIM;
#pragma unroll
      for (int ns = 0; ns < 4; ++ns) {
        int n = n0 + wn * 64 + ns * 16 + lmod;
        hrow[n] = (bf16)gelu_exact(acc[ms][ns][r] + bv[ns]);
      }
    }
  }
}

// ---------------------------------------------------------------------------
// 5) FFN2: out_rows = (h[e] @ W2[e] + b2[e]) * score, atomic scatter-add.
//    M=CAP, N=HDIM, K=FDIM. A tile staged with async global->LDS when
//    available (ASYNCcnt-tracked, bypasses VGPRs).
// ---------------------------------------------------------------------------
__global__ __launch_bounds__(256)
void moe_ffn2_kernel(const bf16* __restrict__ h, const float* __restrict__ W2,
                     const float* __restrict__ b2, const int* __restrict__ idx,
                     const float* __restrict__ scores, float* __restrict__ out) {
  __shared__ bf16 sA[2][MT][LDS_STRIDE];
  __shared__ bf16 sB[2][NT][LDS_STRIDE];

  const int e  = blockIdx.z;
  const int m0 = blockIdx.y * MT;
  const int n0 = blockIdx.x * NT;
  const int tid = threadIdx.x;
  const int lane = tid & 31, wid = tid >> 5;
  const int wm = wid >> 1, wn = wid & 1;
  const int lmod = lane & 15, lhalf = lane >> 4;

  const int ar  = tid >> 1;
  const int akq = (tid & 1) << 4;
  const bf16* aRow = h + ((size_t)e * CAP + m0 + ar) * FDIM;
  const float* w2e = W2 + (size_t)e * FDIM * HDIM;

  v8f acc[2][4];
  v8f zf = {0.f, 0.f, 0.f, 0.f, 0.f, 0.f, 0.f, 0.f};
#pragma unroll
  for (int ms = 0; ms < 2; ++ms)
#pragma unroll
    for (int ns = 0; ns < 4; ++ns) acc[ms][ns] = zf;

  auto stage = [&](int k0, int buf) {
    // A already bf16: async DMA straight into LDS (or plain copy fallback)
#if USE_ASYNC_LDS
    __builtin_amdgcn_global_load_async_to_lds_b128(
        (gvec_p)(aRow + k0 + akq),
        (lvec_p)&sA[buf][ar][akq], 0, 0);
    __builtin_amdgcn_global_load_async_to_lds_b128(
        (gvec_p)(aRow + k0 + akq + 8),
        (lvec_p)&sA[buf][ar][akq + 8], 0, 0);
#else
    u32x4 u0 = *(const u32x4*)(aRow + k0 + akq);
    u32x4 u1 = *(const u32x4*)(aRow + k0 + akq + 8);
    *(u32x4*)&sA[buf][ar][akq]     = u0;
    *(u32x4*)&sA[buf][ar][akq + 8] = u1;
#endif
    // B: fp32 W2 [k][n] -> bf16 LDS transposed
#pragma unroll
    for (int it = 0; it < 4; ++it) {
      int v = tid + it * 256;
      int kk = v >> 5, nv = v & 31;
      f32x4 w = *(const f32x4*)(w2e + (size_t)(k0 + kk) * HDIM + n0 + nv * 4);
#pragma unroll
      for (int j = 0; j < 4; ++j) sB[buf][nv * 4 + j][kk] = (bf16)w[j];
    }
  };

  stage(0, 0);
#if USE_ASYNC_LDS
  __builtin_amdgcn_s_wait_asynccnt(0);
#endif
  __syncthreads();
  const int KSTEPS = FDIM / KT;
  for (int ks = 0; ks < KSTEPS; ++ks) {
    const int buf = ks & 1;
    if (ks + 1 < KSTEPS) stage((ks + 1) * KT, buf ^ 1);
    v16bf af[2], bfr[4];
#pragma unroll
    for (int ms = 0; ms < 2; ++ms)
      af[ms] = load_frag(&sA[buf][wm * 32 + ms * 16 + lmod][lhalf * 8]);
#pragma unroll
    for (int ns = 0; ns < 4; ++ns)
      bfr[ns] = load_frag(&sB[buf][wn * 64 + ns * 16 + lmod][lhalf * 8]);
#pragma unroll
    for (int ms = 0; ms < 2; ++ms)
#pragma unroll
      for (int ns = 0; ns < 4; ++ns)
        acc[ms][ns] = __builtin_amdgcn_wmma_f32_16x16x32_bf16(
            false, af[ms], false, bfr[ns], (short)0, acc[ms][ns], false, false);
#if USE_ASYNC_LDS
    __builtin_amdgcn_s_wait_asynccnt(0);   // next-buffer DMA landed
#endif
    __syncthreads();
  }

  // epilogue: (acc + b2) * score -> atomic scatter-add into final output
  float bv[4];
#pragma unroll
  for (int ns = 0; ns < 4; ++ns)
    bv[ns] = b2[e * HDIM + n0 + wn * 64 + ns * 16 + lmod];
#pragma unroll
  for (int ms = 0; ms < 2; ++ms) {
#pragma unroll
    for (int r = 0; r < 8; ++r) {
      int m = m0 + wm * 32 + ms * 16 + lhalf * 8 + r;
      int tok = idx[e * CAP + m];
      float sc = scores[e * CAP + m];
      float* orow = out + (size_t)tok * HDIM;
#pragma unroll
      for (int ns = 0; ns < 4; ++ns) {
        int n = n0 + wn * 64 + ns * 16 + lmod;
        unsafeAtomicAdd(&orow[n], (acc[ms][ns][r] + bv[ns]) * sc);
      }
    }
  }
}

// ---------------------------------------------------------------------------
// launch
// ---------------------------------------------------------------------------
extern "C" void kernel_launch(void* const* d_in, const int* in_sizes, int n_in,
                              void* d_out, int out_size, void* d_ws, size_t ws_size,
                              hipStream_t stream) {
  const float* x   = (const float*)d_in[0];   // (4,2048,1024)
  const float* gw  = (const float*)d_in[1];   // (1024,8)
  const float* W1  = (const float*)d_in[2];   // (8,1024,4096)
  const float* b1  = (const float*)d_in[3];   // (8,4096)
  const float* W2  = (const float*)d_in[4];   // (8,4096,1024)
  const float* b2  = (const float*)d_in[5];   // (8,1024)
  float* out = (float*)d_out;                 // final (8192*1024) ++ lb_loss (1)

  char* wsb = (char*)d_ws;
  size_t off = 0;
  float* probs   = (float*)(wsb + off); off += (size_t)T_TOK * NEXP * 4;
  int*   idxp    = (int*)  (wsb + off); off += (size_t)NEXP * CAP * 4;
  float* scoresp = (float*)(wsb + off); off += (size_t)NEXP * CAP * 4;
  float* eload   = (float*)(wsb + off); off += 256;
  off = (off + 255) & ~(size_t)255;
  bf16*  hbuf    = (bf16*) (wsb + off);       // 8*1280*4096 bf16 = 80 MB

  // 0) zero scatter-add target
  int n4 = (T_TOK * HDIM) / 4;
  moe_zero_kernel<<<(n4 + 255) / 256, 256, 0, stream>>>((f32x4*)out, n4);
  // 1) gating softmax
  moe_gate_kernel<<<T_TOK / 8, 256, 0, stream>>>(x, gw, probs);
  // 2) per-expert top-capacity selection
  moe_topk_kernel<<<NEXP, 1024, 0, stream>>>(probs, idxp, scoresp, eload);
  // 3) load-balance loss
  moe_lbloss_kernel<<<1, 32, 0, stream>>>(eload, out + (size_t)T_TOK * HDIM);
  // 4) FFN1 (gather + GEMM + GELU)
  dim3 g1(FDIM / NT, CAP / MT, NEXP);   // (32,10,8)
  moe_ffn1_kernel<<<g1, 256, 0, stream>>>(x, W1, b1, idxp, hbuf);
  // 5) FFN2 (GEMM + weighted scatter-add)
  dim3 g2(HDIM / NT, CAP / MT, NEXP);   // (8,10,8)
  moe_ffn2_kernel<<<g2, 256, 0, stream>>>(hbuf, W2, b2, idxp, scoresp, out);
}
